// RNN_71090298683629
// MI455X (gfx1250) — compile-verified, matrix operand order checked by reading, not run on hardware
//
#include <hip/hip_runtime.h>
#include <hip/hip_bf16.h>

#define DEVINL __device__ __forceinline__

typedef __attribute__((ext_vector_type(16))) __bf16 v16bf;
typedef __attribute__((ext_vector_type(8)))  float  v8f;

struct alignas(16) U128 { unsigned x, y, z, w; };
union Frag {
  U128  u[2];
  v16bf v;
};

constexpr int B_  = 16;
constexpr int T_  = 256;
constexpr int E_  = 512;
constexpr int H_  = 1024;
constexpr int H3_ = 3072;
constexpr int V_  = 32000;
constexpr int BT_ = B_ * T_;

constexpr int LDSROW = 40;   // shorts per staged LDS row (32 data + 8 pad)

DEVINL unsigned short f2bf(float f) {
  unsigned u = __float_as_uint(f);
  unsigned r = u + 0x7FFFu + ((u >> 16) & 1u);
  return (unsigned short)(r >> 16);
}

// Load a 16x32 bf16 WMMA fragment: lane (l&15) = row/col index, (l>>4) picks
// which half of K; two 16-byte chunks at k0+hi*8 and k0+16+hi*8 (K-contiguous src).
DEVINL Frag load_frag(const unsigned short* p) {
  Frag f;
  f.u[0] = *(const U128*)p;
  f.u[1] = *(const U128*)(p + 16);
  return f;
}

// gfx1250 async copy: 16B global -> LDS, tracked by ASYNCcnt.
DEVINL void async_ld16(const unsigned short* gptr, unsigned lds_byte_off) {
  asm volatile("global_load_async_to_lds_b128 %0, %1, off"
               :: "v"(lds_byte_off), "v"(gptr) : "memory");
}
DEVINL void wait_async0() {
  asm volatile("s_wait_asynccnt 0x0" ::: "memory");
}

// ---------------------------------------------------------------- f32 -> bf16
__global__ void k_cvt(const float* __restrict__ s, unsigned short* __restrict__ d, int n) {
  int i = blockIdx.x * blockDim.x + threadIdx.x;
  if (i < n) d[i] = f2bf(s[i]);
}

// ---------------------------------------------------- embedding gather + cvt
__global__ void k_embed(const int* __restrict__ idx, const float* __restrict__ emb,
                        unsigned short* __restrict__ x) {
  int i = blockIdx.x * blockDim.x + threadIdx.x;   // over BT*E
  if (i >= BT_ * E_) return;
  int bt = i >> 9;            // /E_
  int e  = i & (E_ - 1);
  x[i] = f2bf(emb[(size_t)idx[bt] * E_ + e]);
}

// -------------------------------------------------- zero h buffers + barrier
__global__ void k_init(float* hf, unsigned short* hb, unsigned* sync) {
  int i = blockIdx.x * blockDim.x + threadIdx.x;
  if (i < 2 * B_ * H_) { hf[i] = 0.f; hb[i] = 0; }
  if (i == 0) sync[0] = 0u;
}

// ------------------------------------------- WMMA GEMM, direct global loads
// C[M,N] = A_bf16[M,K] * W_bf16[N,K]^T + bias[N]
// 256 threads = 8 waves; block tile 128x128; wave tile 64x32 (4x2 WMMA tiles).
__global__ __launch_bounds__(256)
void k_gemm_bf16(const unsigned short* __restrict__ A,
                 const unsigned short* __restrict__ W,
                 const float* __restrict__ bias,
                 float* __restrict__ Cf,
                 unsigned short* __restrict__ Cbf,
                 int M, int N, int K, int relu)
{
  const int lane = threadIdx.x & 31;
  const int wv   = threadIdx.x >> 5;
  const int wr   = wv >> 2;          // 0..1
  const int wc   = wv & 3;           // 0..3
  const int ml   = lane & 15;
  const int hi   = lane >> 4;
  const int mbase = blockIdx.y * 128 + wr * 64;
  const int nbase = blockIdx.x * 128 + wc * 32;

  v8f acc[4][2] = {};

  for (int k0 = 0; k0 < K; k0 += 32) {
    Frag a[4], b[2];
#pragma unroll
    for (int i = 0; i < 4; i++)
      a[i] = load_frag(A + (size_t)(mbase + i * 16 + ml) * K + k0 + hi * 8);
#pragma unroll
    for (int j = 0; j < 2; j++)
      b[j] = load_frag(W + (size_t)(nbase + j * 16 + ml) * K + k0 + hi * 8);
#pragma unroll
    for (int i = 0; i < 4; i++)
#pragma unroll
      for (int j = 0; j < 2; j++)
        acc[i][j] = __builtin_amdgcn_wmma_f32_16x16x32_bf16(
            false, a[i].v, false, b[j].v, (short)0, acc[i][j], false, false);
  }

#pragma unroll
  for (int i = 0; i < 4; i++) {
#pragma unroll
    for (int j = 0; j < 2; j++) {
      const int col = nbase + j * 16 + ml;
      const float bb = bias[col];
#pragma unroll
      for (int v = 0; v < 8; v++) {
        const int row = mbase + i * 16 + v + 8 * hi;
        float x = acc[i][j][v] + bb;
        if (relu) x = fmaxf(x, 0.f);
        if (Cf)  Cf [(size_t)row * N + col] = x;
        if (Cbf) Cbf[(size_t)row * N + col] = f2bf(x);
      }
    }
  }
}

// --------------------- WMMA GEMM, async-to-LDS staged, double-buffered
// Same math/tiling as k_gemm_bf16, but A/B 128x32 tiles are staged into LDS
// with global_load_async_to_lds_b128 (ASYNCcnt) and fragments served from LDS.
__global__ __launch_bounds__(256)
void k_gemm_bf16_lds(const unsigned short* __restrict__ A,
                     const unsigned short* __restrict__ W,
                     const float* __restrict__ bias,
                     float* __restrict__ Cf,
                     unsigned short* __restrict__ Cbf,
                     int M, int N, int K, int relu)
{
  __shared__ unsigned short smem[4 * 128 * LDSROW];   // A0 B0 A1 B1

  const int tid  = threadIdx.x;
  const int lane = tid & 31;
  const int wv   = tid >> 5;
  const int wr   = wv >> 2;          // 0..1
  const int wc   = wv & 3;           // 0..3
  const int ml   = lane & 15;
  const int hi   = lane >> 4;
  const int mbase = blockIdx.y * 128;
  const int nbase = blockIdx.x * 128;

  // staging map: thread -> (row = tid>>1, half = (tid&1)*16 shorts), 2x16B each
  const int crow  = tid >> 1;
  const int chalf = (tid & 1) * 16;
  const unsigned short* gA = A + (size_t)(mbase + crow) * K + chalf;
  const unsigned short* gW = W + (size_t)(nbase + crow) * K + chalf;

  const unsigned smem_base = (unsigned)(uintptr_t)smem;           // LDS byte offset
  const unsigned rowOff    = (unsigned)(crow * LDSROW + chalf) * 2;
  const unsigned bufBytes  = 128 * LDSROW * 2;

  v8f acc[4][2] = {};

  // prologue: stage k0 = 0 into buffer 0
  {
    const unsigned aoff = smem_base + rowOff;
    const unsigned boff = smem_base + bufBytes + rowOff;
    async_ld16(gA,     aoff);
    async_ld16(gA + 8, aoff + 16);
    async_ld16(gW,     boff);
    async_ld16(gW + 8, boff + 16);
  }
  wait_async0();
  __syncthreads();

  int buf = 0;
  for (int k0 = 0; k0 < K; k0 += 32, buf ^= 1) {
    if (k0 + 32 < K) {                 // stage next K-slab into other buffer
      const unsigned aoff = smem_base + (unsigned)(buf ^ 1) * 2 * bufBytes + rowOff;
      const unsigned boff = aoff + bufBytes;
      async_ld16(gA + k0 + 32,     aoff);
      async_ld16(gA + k0 + 40,     aoff + 16);
      async_ld16(gW + k0 + 32,     boff);
      async_ld16(gW + k0 + 40,     boff + 16);
    }

    const unsigned short* sA = smem + buf * 2 * 128 * LDSROW;
    const unsigned short* sB = sA + 128 * LDSROW;

    Frag a[4], b[2];
#pragma unroll
    for (int i = 0; i < 4; i++)
      a[i] = load_frag(sA + (wr * 64 + i * 16 + ml) * LDSROW + hi * 8);
#pragma unroll
    for (int j = 0; j < 2; j++)
      b[j] = load_frag(sB + (wc * 32 + j * 16 + ml) * LDSROW + hi * 8);
#pragma unroll
    for (int i = 0; i < 4; i++)
#pragma unroll
      for (int j = 0; j < 2; j++)
        acc[i][j] = __builtin_amdgcn_wmma_f32_16x16x32_bf16(
            false, a[i].v, false, b[j].v, (short)0, acc[i][j], false, false);

    wait_async0();                      // next slab landed (all lanes of this WG
    __syncthreads();                    // observe it after the barrier)
  }

#pragma unroll
  for (int i = 0; i < 4; i++) {
#pragma unroll
    for (int j = 0; j < 2; j++) {
      const int col = nbase + wc * 32 + j * 16 + ml;
      const float bb = bias[col];
#pragma unroll
      for (int v = 0; v < 8; v++) {
        const int row = mbase + wr * 64 + i * 16 + v + 8 * hi;
        float x = acc[i][j][v] + bb;
        if (relu) x = fmaxf(x, 0.f);
        if (Cf)  Cf [(size_t)row * N + col] = x;
        if (Cbf) Cbf[(size_t)row * N + col] = f2bf(x);
      }
    }
  }
}

// -------------------------------------------------------- GRU recurrence
// 16 persistent workgroups x 128 threads (4 waves). Wave owns 16 h-columns,
// computes the r/z/n 16x16 gate tiles per step via WMMA (K=H), then the gate
// math, double-buffers h (f32 + bf16) in global, grid-syncs via atomic phase.
__global__ __launch_bounds__(128)
void k_gru(const float* __restrict__ gi,          // [B,T,3H], includes b_ih
           const unsigned short* __restrict__ Whh, // [3H,H] bf16
           const float* __restrict__ b_hh,          // [3H]
           float* __restrict__ h_f32,               // 2 * [B,H]
           unsigned short* __restrict__ h_bf,       // 2 * [B,H]
           unsigned short* __restrict__ hs_bf,      // [B,T,H]
           float* __restrict__ hidden,              // [B,H] (into d_out tail)
           unsigned* __restrict__ sync)
{
  const int lane = threadIdx.x & 31;
  const int wv   = threadIdx.x >> 5;                 // 0..3
  const int colBase = blockIdx.x * 64 + wv * 16;     // h column tile
  const int ml = lane & 15;
  const int hi = lane >> 4;
  const int col = colBase + ml;

  const float bhr = b_hh[col];
  const float bhz = b_hh[H_ + col];
  const float bhn = b_hh[2 * H_ + col];

  for (int t = 0; t < T_; t++) {
    const int rd = t & 1, wb = rd ^ 1;
    const unsigned short* hb = h_bf + (size_t)rd * B_ * H_;

    v8f acc[3] = {};
    for (int k0 = 0; k0 < H_; k0 += 32) {
      Frag a = load_frag(hb + (size_t)ml * H_ + k0 + hi * 8);
#pragma unroll
      for (int g = 0; g < 3; g++) {
        Frag b = load_frag(Whh + (size_t)(g * H_ + colBase + ml) * H_ + k0 + hi * 8);
        acc[g] = __builtin_amdgcn_wmma_f32_16x16x32_bf16(
            false, a.v, false, b.v, (short)0, acc[g], false, false);
      }
    }

#pragma unroll
    for (int v = 0; v < 8; v++) {
      const int b = v + 8 * hi;                       // batch row
      const size_t gbase = (size_t)(b * T_ + t) * H3_;
      const float gir = gi[gbase + col];
      const float giz = gi[gbase + H_ + col];
      const float gin = gi[gbase + 2 * H_ + col];
      if (t + 1 < T_)                                 // global_prefetch next step
        __builtin_prefetch(gi + gbase + H3_ + col, 0, 0);
      const float ghr = acc[0][v] + bhr;
      const float ghz = acc[1][v] + bhz;
      const float ghn = acc[2][v] + bhn;
      const float r = 1.f / (1.f + __expf(-(gir + ghr)));
      const float z = 1.f / (1.f + __expf(-(giz + ghz)));
      const float n = tanhf(gin + r * ghn);
      const float hp = h_f32[(size_t)rd * B_ * H_ + (size_t)b * H_ + col];
      const float hn = (1.f - z) * n + z * hp;
      h_f32[(size_t)wb * B_ * H_ + (size_t)b * H_ + col] = hn;
      h_bf [(size_t)wb * B_ * H_ + (size_t)b * H_ + col] = f2bf(hn);
      hs_bf[(size_t)(b * T_ + t) * H_ + col] = f2bf(hn);
      if (t == T_ - 1) hidden[(size_t)b * H_ + col] = hn;
    }

    // grid-wide phase barrier
    __threadfence();
    __syncthreads();
    if (threadIdx.x == 0) {
      atomicAdd(sync, 1u);
      const unsigned target = (unsigned)(t + 1) * gridDim.x;
      while (atomicAdd(sync, 0u) < target) __builtin_amdgcn_s_sleep(1);
    }
    __syncthreads();
  }
}

// ------------------------------------------------------------ log_softmax
__global__ __launch_bounds__(256)
void k_logsoftmax(float* __restrict__ out, int N)
{
  __shared__ float red[256];
  float* p = out + (size_t)blockIdx.x * N;
  const int tid = threadIdx.x;

  float mx = -3.402823466e+38f;
  for (int i = tid; i < N; i += 256) mx = fmaxf(mx, p[i]);
  red[tid] = mx; __syncthreads();
  for (int s = 128; s > 0; s >>= 1) {
    if (tid < s) red[tid] = fmaxf(red[tid], red[tid + s]);
    __syncthreads();
  }
  mx = red[0]; __syncthreads();

  float sum = 0.f;
  for (int i = tid; i < N; i += 256) sum += __expf(p[i] - mx);
  red[tid] = sum; __syncthreads();
  for (int s = 128; s > 0; s >>= 1) {
    if (tid < s) red[tid] += red[tid + s];
    __syncthreads();
  }
  const float lse = __logf(red[0]) + mx;
  __syncthreads();

  for (int i = tid; i < N; i += 256) p[i] = p[i] - lse;
}

// ---------------------------------------------------------------- launcher
extern "C" void kernel_launch(void* const* d_in, const int* in_sizes, int n_in,
                              void* d_out, int out_size, void* d_ws, size_t ws_size,
                              hipStream_t stream)
{
  const int*   inputs = (const int*)  d_in[0];
  const float* emb    = (const float*)d_in[1];
  const float* W_ih   = (const float*)d_in[2];
  const float* W_hh   = (const float*)d_in[3];
  const float* b_ih   = (const float*)d_in[4];
  const float* b_hh   = (const float*)d_in[5];
  const float* W1     = (const float*)d_in[6];
  const float* b1     = (const float*)d_in[7];
  const float* W2     = (const float*)d_in[8];
  const float* b2     = (const float*)d_in[9];
  float* out = (float*)d_out;

  char* ws = (char*)d_ws;
  size_t o = 0;
  auto carve = [&](size_t bytes) { size_t r = o; o += (bytes + 255) & ~(size_t)255; return r; };

  unsigned short* Wih_bf = (unsigned short*)(ws + carve((size_t)H3_ * E_ * 2));
  unsigned short* Whh_bf = (unsigned short*)(ws + carve((size_t)H3_ * H_ * 2));
  unsigned short* W1_bf  = (unsigned short*)(ws + carve((size_t)H_  * H_ * 2));
  unsigned short* W2_bf  = (unsigned short*)(ws + carve((size_t)V_  * H_ * 2));
  unsigned short* x_bf   = (unsigned short*)(ws + carve((size_t)BT_ * E_ * 2));
  float*          gi     = (float*)         (ws + carve((size_t)BT_ * H3_ * 4));
  unsigned short* hs_bf  = (unsigned short*)(ws + carve((size_t)BT_ * H_ * 2));
  unsigned short* y_bf   = (unsigned short*)(ws + carve((size_t)BT_ * H_ * 2));
  float*          h_f32  = (float*)         (ws + carve((size_t)2 * B_ * H_ * 4));
  unsigned short* h_bf   = (unsigned short*)(ws + carve((size_t)2 * B_ * H_ * 2));
  unsigned*       sync   = (unsigned*)      (ws + carve(256));

  // 1) weights -> bf16
  {
    int n;
    n = H3_ * E_; hipLaunchKernelGGL(k_cvt, dim3((n + 255) / 256), dim3(256), 0, stream, W_ih, Wih_bf, n);
    n = H3_ * H_; hipLaunchKernelGGL(k_cvt, dim3((n + 255) / 256), dim3(256), 0, stream, W_hh, Whh_bf, n);
    n = H_ * H_;  hipLaunchKernelGGL(k_cvt, dim3((n + 255) / 256), dim3(256), 0, stream, W1,   W1_bf,  n);
    n = V_ * H_;  hipLaunchKernelGGL(k_cvt, dim3((n + 255) / 256), dim3(256), 0, stream, W2,   W2_bf,  n);
  }

  // 2) embedding gather -> bf16 x
  hipLaunchKernelGGL(k_embed, dim3((BT_ * E_ + 255) / 256), dim3(256), 0, stream, inputs, emb, x_bf);

  // 3) gi = x * W_ih^T + b_ih   (M=4096, N=3072, K=512) -> f32   [async-LDS staged]
  hipLaunchKernelGGL(k_gemm_bf16_lds, dim3(H3_ / 128, BT_ / 128), dim3(256), 0, stream,
                     x_bf, Wih_bf, b_ih, gi, (unsigned short*)nullptr, BT_, H3_, E_, 0);

  // 4) init h double-buffers + barrier counter
  hipLaunchKernelGGL(k_init, dim3((2 * B_ * H_ + 255) / 256), dim3(256), 0, stream, h_f32, h_bf, sync);

  // 5) GRU scan (writes hs_bf and hidden tail of d_out)
  hipLaunchKernelGGL(k_gru, dim3(16), dim3(128), 0, stream,
                     gi, Whh_bf, b_hh, h_f32, h_bf, hs_bf, out + (size_t)BT_ * V_, sync);

  // 6) y = relu(hs * W1^T + b1) -> bf16   (M=4096, N=1024, K=1024)  [direct]
  hipLaunchKernelGGL(k_gemm_bf16, dim3(H_ / 128, BT_ / 128), dim3(256), 0, stream,
                     hs_bf, W1_bf, b1, (float*)nullptr, y_bf, BT_, H_, H_, 1);

  // 7) logits = y * W2^T + b2 -> d_out    (M=4096, N=32000, K=1024) [async-LDS staged]
  hipLaunchKernelGGL(k_gemm_bf16_lds, dim3(V_ / 128, BT_ / 128), dim3(256), 0, stream,
                     y_bf, W2_bf, b2, out, (unsigned short*)nullptr, BT_, V_, H_, 0);

  // 8) in-place log_softmax over V for each of the 4096 rows
  hipLaunchKernelGGL(k_logsoftmax, dim3(BT_), dim3(256), 0, stream, out, V_);
}
